// HybridSTNDelayLineRNN_4286377361420
// MI455X (gfx1250) — matrix operand order repaired and checked by reading, not compiled, not run on hardware
//
#include <hip/hip_runtime.h>
#include <math.h>

typedef __attribute__((ext_vector_type(16))) __bf16 v16bf;
typedef __attribute__((ext_vector_type(8)))  float  v8f;

namespace {
constexpr int   kT     = 16384;
constexpr int   kB     = 64;
constexpr int   kH     = 64;
constexpr float kBlend = 1.0f / 1.5f;   // K = delay / os_factor
}

// Branch-free tanh: gfx1250 V_TANH_F32 if the builtin exists, else a straight-line
// exp2/rcp formulation (no exec-mask branches on the serial critical path).
static __device__ __forceinline__ float fast_tanh(float v) {
#if __has_builtin(__builtin_amdgcn_tanhf)
  return __builtin_amdgcn_tanhf(v);
#else
  const float xc = fminf(fmaxf(v, -9.0f), 9.0f);      // tanh(+-9) rounds to +-1 in f32
  const float e  = __builtin_amdgcn_exp2f(xc * 2.885390081777927f);  // 2*log2(e)
  return (e - 1.0f) * __builtin_amdgcn_rcpf(e + 1.0f);
#endif
}

// One wave32 per 16-row batch tile. Weight-stationary WMMA:
//   D[m(outH) x n(batch)] = A(W_hh tile 16x64, bf16) * B(h state 64x16, bf16) + C(xw + biases)
// A preloaded once; B rebuilt each step from the f32 state kept in D layout
// (bf16 pair-pack -> 16 dword shuffles -> 16 selects per step).
__launch_bounds__(32)
__global__ void rnn_delayline_scan(const float* __restrict__ x,      // [B][T]  (I==1)
                                   const float* __restrict__ W_ih,   // [H]     (I==1)
                                   const float* __restrict__ W_hh,   // [H][H]
                                   const float* __restrict__ b_ih,   // [H]
                                   const float* __restrict__ b_hh,   // [H]
                                   float* __restrict__ states,       // [B][T][H]
                                   float* __restrict__ h_last)       // [B][H]
{
  const int  lane  = threadIdx.x;       // 0..31
  const int  lo    = lane & 15;
  const int  hi    = lane >> 4;         // half-wave id
  const bool upper = (hi != 0);
  const int  bRow  = blockIdx.x * 16 + lo;   // batch row on the WMMA N axis

  // ---- Preload W_hh into WMMA A layout (4 M-tiles x 2 K-chunks), bf16 ----
  // A 16x32 (contiguous k order used here, same as B): both half-waves hold M=lo; halves i=0..15 -> K = 16*hi + i;
  // second chunk offsets K by 32.
  v16bf A[4][2];
  #pragma unroll
  for (int j = 0; j < 4; ++j) {
    const int m = 16 * j + lo;
    #pragma unroll
    for (int c = 0; c < 2; ++c) {
      #pragma unroll
      for (int i = 0; i < 16; ++i) {
        const int k = 32 * c + 16 * hi + i;
        A[j][c][i] = (__bf16)W_hh[m * kH + k];
      }
    }
  }

  // ---- Loop-invariant input-projection constants (D layout: m = 16j + 8*hi + v) ----
  float wih[4][8], bsum[4][8];
  #pragma unroll
  for (int j = 0; j < 4; ++j)
    #pragma unroll
    for (int v = 0; v < 8; ++v) {
      const int m = 16 * j + 8 * hi + v;
      wih[j][v]  = W_ih[m];
      bsum[j][v] = b_ih[m] + b_hh[m];
    }

  // ---- Recurrent state in f32, kept in the WMMA D layout ----
  float h[4][8];
  #pragma unroll
  for (int j = 0; j < 4; ++j)
    #pragma unroll
    for (int v = 0; v < 8; ++v) h[j][v] = 0.0f;

  v16bf B0 = {};   // h == 0 at t == 0
  v16bf B1 = {};

  const float* xrow = x + (size_t)bRow * kT;
  float*       srow = states + (size_t)bRow * kT * kH;

  float xv = xrow[0];   // software-pipelined input scalar

  #pragma unroll 1
  for (int t = 0; t < kT; ++t) {
    const int   tn = (t + 1 < kT) ? (t + 1) : (kT - 1);
    const float xn = xrow[tn];                     // prefetch next step's x

    // C init: xw_t + b_ih + b_hh
    v8f cacc[4];
    #pragma unroll
    for (int j = 0; j < 4; ++j)
      #pragma unroll
      for (int v = 0; v < 8; ++v)
        cacc[j][v] = fmaf(xv, wih[j][v], bsum[j][v]);

    // D = A*B + C : 4 independent accumulator chains, 2 chained K-steps each
    #pragma unroll
    for (int j = 0; j < 4; ++j) {
      cacc[j] = __builtin_amdgcn_wmma_f32_16x16x32_bf16(
          false, A[j][0], false, B0, (short)0, cacc[j], false, false);
      cacc[j] = __builtin_amdgcn_wmma_f32_16x16x32_bf16(
          false, A[j][1], false, B1, (short)0, cacc[j], false, false);
    }

    // tanh + fractional-delay blend (branch-free), state update
    #pragma unroll
    for (int j = 0; j < 4; ++j)
      #pragma unroll
      for (int v = 0; v < 8; ++v) {
        const float cell = fast_tanh(cacc[j][v]);
        h[j][v] = h[j][v] + kBlend * (cell - h[j][v]);
      }

    // Streamed store of states[b][t][:]: two b128 per tile, 16B-aligned.
    float* sp = srow + (size_t)t * kH + 8 * hi;
    #pragma unroll
    for (int j = 0; j < 4; ++j) {
      *(float4*)(sp + 16 * j)     = make_float4(h[j][0], h[j][1], h[j][2], h[j][3]);
      *(float4*)(sp + 16 * j + 4) = make_float4(h[j][4], h[j][5], h[j][6], h[j][7]);
    }

    // ---- Rebuild B operands for the next step ----
    // Pack each tile into bf16 pairs (4 dwords), shuffle dwords across half-waves
    // (lane ^ 16), then select per half. B k order used here (matches A): half i of the
    // operand = K offset 16*hi + i; D holds m = 16j + 8*hi + v.
    unsigned pk[4][4];
    #pragma unroll
    for (int j = 0; j < 4; ++j)
      #pragma unroll
      for (int w = 0; w < 4; ++w) {
        union { unsigned u; __bf16 b[2]; } p;
        p.b[0] = (__bf16)h[j][2 * w];
        p.b[1] = (__bf16)h[j][2 * w + 1];
        pk[j][w] = p.u;
      }
    unsigned xpk[4][4];
    #pragma unroll
    for (int j = 0; j < 4; ++j)
      #pragma unroll
      for (int w = 0; w < 4; ++w)
        xpk[j][w] = (unsigned)__shfl_xor((int)pk[j][w], 16, 32);

    union { v16bf v; unsigned u[8]; } nb0, nb1;
    #pragma unroll
    for (int w = 0; w < 4; ++w) {
      // B0 halves 0..7  : lower lanes m=0..7  (own t0) | upper lanes m=16..23 (partner t1)
      // B0 halves 8..15 : lower lanes m=8..15 (partner t0) | upper lanes m=24..31 (own t1)
      nb0.u[w]     = upper ? xpk[1][w] : pk[0][w];
      nb0.u[4 + w] = upper ? pk[1][w]  : xpk[0][w];
      // B1: same pattern over h rows m=32..63 (tiles 2,3)
      nb1.u[w]     = upper ? xpk[3][w] : pk[2][w];
      nb1.u[4 + w] = upper ? pk[3][w]  : xpk[2][w];
    }
    B0 = nb0.v;
    B1 = nb1.v;

    xv = xn;
  }

  // ---- Final hidden state -> h_last[b][:] ----
  #pragma unroll
  for (int j = 0; j < 4; ++j)
    #pragma unroll
    for (int v = 0; v < 8; ++v)
      h_last[(size_t)bRow * kH + 16 * j + 8 * hi + v] = h[j][v];
}

extern "C" void kernel_launch(void* const* d_in, const int* in_sizes, int n_in,
                              void* d_out, int out_size, void* d_ws, size_t ws_size,
                              hipStream_t stream) {
  (void)in_sizes; (void)n_in; (void)out_size; (void)d_ws; (void)ws_size;
  const float* x    = (const float*)d_in[0];
  const float* W_ih = (const float*)d_in[1];
  const float* W_hh = (const float*)d_in[2];
  const float* b_ih = (const float*)d_in[3];
  const float* b_hh = (const float*)d_in[4];

  float* states = (float*)d_out;                               // [B][T][H]
  float* h_last = (float*)d_out + (size_t)kB * kT * kH;        // [1][B][H]

  rnn_delayline_scan<<<kB / 16, 32, 0, stream>>>(x, W_ih, W_hh, b_ih, b_hh,
                                                 states, h_last);
}
